// DiffCLIP_16973710754150
// MI455X (gfx1250) — matrix operand order repaired and checked by reading, not compiled, not run on hardware
//
#include <hip/hip_runtime.h>

typedef _Float16 f16;
typedef __attribute__((ext_vector_type(16))) _Float16 v16h;
typedef __attribute__((ext_vector_type(8)))  float    v8f;

#define LSEQ 1024
#define NB 4
#define DM 1024
#define NH 16
#define HDIM 64
#define HE 8
#define LAMBDA_INIT 0.8f

union F16Frag { v16h v; uint4 q[2]; };

// ---------------------------------------------------------------------------
// gfx1250 async copy: 32 bytes global -> LDS per lane (2 x b128, ASYNCcnt).
// LDS byte address = low 32 bits of the generic pointer (aperture layout).
// ---------------------------------------------------------------------------
__device__ __forceinline__ void async_copy32(const f16* __restrict__ g, f16* l) {
  const unsigned lds = (unsigned)(uintptr_t)l;
  asm volatile(
      "global_load_async_to_lds_b128 %0, %1, off\n\t"
      "global_load_async_to_lds_b128 %0, %1, off offset:16"
      :: "v"(lds), "v"(g) : "memory");
}

__device__ __forceinline__ void wait_async0() {
#if __has_builtin(__builtin_amdgcn_s_wait_asynccnt)
  __builtin_amdgcn_s_wait_asynccnt(0);
#else
  asm volatile("s_wait_asynccnt 0x0" ::: "memory");
#endif
}

// ---------------------------------------------------------------------------
// x = swapaxes(query,0,1) converted to f16, row-major (B*L, D)
// ---------------------------------------------------------------------------
__global__ void cvt_x_kernel(const float* __restrict__ query, f16* __restrict__ x16) {
  const int idx = blockIdx.x * blockDim.x + threadIdx.x;   // over 4M elements
  const int d = idx & (DM - 1);
  const int m = idx >> 10;            // m = b*L + l
  const int b = m >> 10, l = m & (LSEQ - 1);
  x16[idx] = (f16)query[((size_t)l * NB + b) * DM + d];
}

// ---------------------------------------------------------------------------
// Weight convert + transpose: W[k][n] f32 -> WT[n][k] f16 (LDS-tiled)
// ---------------------------------------------------------------------------
__global__ void cvt_wT_kernel(const float* __restrict__ W, f16* __restrict__ WT) {
  __shared__ float tile[32][33];
  const int tx = threadIdx.x, ty = threadIdx.y;           // 32 x 8
  const int x0 = blockIdx.x * 32, y0 = blockIdx.y * 32;
#pragma unroll
  for (int i = 0; i < 4; ++i)
    tile[ty + i * 8][tx] = W[(size_t)(y0 + ty + i * 8) * DM + (x0 + tx)];
  __syncthreads();
#pragma unroll
  for (int i = 0; i < 4; ++i)
    WT[(size_t)(x0 + ty + i * 8) * DM + (y0 + tx)] = (f16)tile[tx][ty + i * 8];
}

// ---------------------------------------------------------------------------
// lambda = exp(sum lq1*lk1) - exp(sum lq2*lk2) + 0.8
// ---------------------------------------------------------------------------
__global__ void lambda_kernel(const float* __restrict__ lq1, const float* __restrict__ lk1,
                              const float* __restrict__ lq2, const float* __restrict__ lk2,
                              float* __restrict__ lam) {
  __shared__ float s1[64], s2[64];
  const int t = threadIdx.x;
  s1[t] = lq1[t] * lk1[t];
  s2[t] = lq2[t] * lk2[t];
  __syncthreads();
  if (t == 0) {
    float a = 0.f, b = 0.f;
    for (int i = 0; i < 64; ++i) { a += s1[i]; b += s2[i]; }
    *lam = __expf(a) - __expf(b) + LAMBDA_INIT;
  }
}

// ---------------------------------------------------------------------------
// WMMA GEMM: C(4096 x 1024) = A_f16 (4096 x 1024) x WT_f16 (1024 x 1024)^T + bias
// Block: 128 threads (4 waves), tile 64(M) x 64(N), K-step 32.
// Double-buffered LDS staging via gfx1250 async global->LDS copies:
// one barrier per K-step; the fill for step k+1 overlaps the WMMAs of step k.
// MODE: 0=Q (scale, head-major), 1=K (head-major), 2=V (per-head transposed),
//       3=O (f32, (L,N,D) order)
// ---------------------------------------------------------------------------
template <int MODE>
__global__ void gemm_kernel(const f16* __restrict__ A, const f16* __restrict__ WT,
                            const float* __restrict__ bias, void* __restrict__ out) {
  __shared__ f16 At[2][64 * 32];
  __shared__ f16 Bt[2][64 * 32];
  const int t = threadIdx.x;
  const int lane = t & 31, wv = t >> 5;
  const int m0 = blockIdx.y * 64;
  const int n0 = blockIdx.x * 64;
  v8f acc[4] = {};

  const int srow = t >> 1, scol = (t & 1) * 16;           // staging: 16 halves/thread
  const int arow = wv * 16 + (lane & 15);
  const int ko = (lane < 16) ? 0 : 8;                     // A-frag K offset
  const int kb = (lane < 16) ? 0 : 8;                     // B-frag K offset

  const f16* gA = A + (size_t)(m0 + srow) * DM + scol;
  const f16* gB = WT + (size_t)(n0 + srow) * DM + scol;
  f16* lA = &At[0][0] + srow * 32 + scol;
  f16* lB = &Bt[0][0] + srow * 32 + scol;

  // prologue: fill buffer 0
  async_copy32(gA, lA);
  async_copy32(gB, lB);

  for (int ks = 0; ks < 32; ++ks) {
    const int cur = ks & 1;
    wait_async0();        // own async fills complete
    __syncthreads();      // (includes s_wait_dscnt 0) -> all waves' fills/reads ordered
    if (ks + 1 < 32) {    // overlap next fill with this step's WMMAs
      const int nk = (ks + 1) * 32;
      async_copy32(gA + nk, lA + (cur ^ 1) * (64 * 32));
      async_copy32(gB + nk, lB + (cur ^ 1) * (64 * 32));
    }

    F16Frag fa;
    fa.q[0] = *(const uint4*)(&At[cur][0] + arow * 32 + ko);
    fa.q[1] = *(const uint4*)(&At[cur][0] + arow * 32 + ko + 16);
#pragma unroll
    for (int j = 0; j < 4; ++j) {
      F16Frag fb;
      const int bcol = j * 16 + (lane & 15);
      fb.q[0] = *(const uint4*)(&Bt[cur][0] + bcol * 32 + kb);
      fb.q[1] = *(const uint4*)(&Bt[cur][0] + bcol * 32 + kb + 16);
      acc[j] = __builtin_amdgcn_wmma_f32_16x16x32_f16(
          false, fa.v, false, fb.v, (short)0, acc[j], false, false);
    }
  }

#pragma unroll
  for (int j = 0; j < 4; ++j) {
#pragma unroll
    for (int r = 0; r < 8; ++r) {
      const int M = m0 + wv * 16 + r + ((lane >= 16) ? 8 : 0);
      const int N = n0 + j * 16 + (lane & 15);
      const float val = acc[j][r] + bias[N];
      const int b = M >> 10, ll = M & (LSEQ - 1);
      if (MODE == 0) {        // Q head-major (B,H,L,HD), scaled by HD^-0.5
        ((f16*)out)[(((size_t)b * NH + (N >> 6)) * LSEQ + ll) * HDIM + (N & 63)] =
            (f16)(val * 0.125f);
      } else if (MODE == 1) { // K head-major (B,H,L,HD)
        ((f16*)out)[(((size_t)b * NH + (N >> 6)) * LSEQ + ll) * HDIM + (N & 63)] = (f16)val;
      } else if (MODE == 2) { // V transposed per head-pair: (B,HE,128,L)
        ((f16*)out)[(((size_t)b * HE + (N >> 7)) * 128 + (N & 127)) * LSEQ + ll] = (f16)val;
      } else {                // O: f32 output in (L,N,D)
        ((float*)out)[((size_t)ll * NB + b) * DM + N] = val;
      }
    }
  }
}

// ---------------------------------------------------------------------------
// Differential attention: one block per (b, head-pair, 16-row q tile).
// 128 threads = 4 waves. Dynamic LDS: scores f32 [16x1024], diff f16 [16x1024],
// out f32 [16x128], rstd f32 [16].
// ---------------------------------------------------------------------------
__global__ void attn_kernel(const f16* __restrict__ q16, const f16* __restrict__ k16,
                            const f16* __restrict__ vt16, const float* __restrict__ mask,
                            const float* __restrict__ lamPtr, const float* __restrict__ normw,
                            f16* __restrict__ attn16) {
  extern __shared__ char smem[];
  float* sS = (float*)smem;                                   // 64 KB
  f16*   sD = (f16*)(smem + 16 * 1024 * 4);                   // 32 KB
  float* sO = (float*)(smem + 16 * 1024 * 4 + 16 * 1024 * 2); // 8 KB
  float* sR = sO + 16 * 128;                                  // 16 floats

  const int t = threadIdx.x;           // 0..127
  const int lane = t & 31, wv = t >> 5;
  const int bid = blockIdx.x;
  const int qt = bid & 63;
  const int he = (bid >> 6) & 7;
  const int b  = bid >> 9;
  const int q0 = qt * 16;
  const float lam = lamPtr[0];

  for (int hi = 0; hi < 2; ++hi) {
    const int bh = b * NH + he * 2 + hi;
    const f16* qh = q16 + (size_t)bh * LSEQ * HDIM;
    const f16* kh = k16 + (size_t)bh * LSEQ * HDIM;

    // q A-fragments for both 32-wide K steps over HD=64
    F16Frag fq0, fq1;
    {
      const int row = q0 + (lane & 15);
      const int ko = (lane < 16) ? 0 : 8;
      const f16* base = qh + (size_t)row * HDIM;
      fq0.q[0] = *(const uint4*)(base + ko);
      fq0.q[1] = *(const uint4*)(base + ko + 16);
      fq1.q[0] = *(const uint4*)(base + 32 + ko);
      fq1.q[1] = *(const uint4*)(base + 32 + ko + 16);
    }
    // scores 16 x 1024: each wave covers 16 column tiles
    for (int tile = wv; tile < 64; tile += 4) {
      const int n0 = tile * 16;
      v8f acc = {};
      const int col = n0 + (lane & 15);
      const int kbb = (lane < 16) ? 0 : 8;
      const f16* kbase = kh + (size_t)col * HDIM + kbb;
      F16Frag fb;
      fb.q[0] = *(const uint4*)(kbase);
      fb.q[1] = *(const uint4*)(kbase + 16);
      acc = __builtin_amdgcn_wmma_f32_16x16x32_f16(false, fq0.v, false, fb.v, (short)0, acc, false, false);
      fb.q[0] = *(const uint4*)(kbase + 32);
      fb.q[1] = *(const uint4*)(kbase + 48);
      acc = __builtin_amdgcn_wmma_f32_16x16x32_f16(false, fq1.v, false, fb.v, (short)0, acc, false, false);
#pragma unroll
      for (int r = 0; r < 8; ++r) {
        const int M = r + ((lane >= 16) ? 8 : 0);
        sS[M * LSEQ + col] = acc[r];
      }
    }
    __syncthreads();
    // full-row softmax; 8 lanes per row (rows of one wave stay in that wave)
    {
      const int row = t >> 3;
      const int seg = t & 7;
      const float* mrow = mask + (size_t)(q0 + row) * LSEQ;
      float m = -3.4e38f;
      for (int i = 0; i < 128; ++i) {
        const int col = seg * 128 + i;
        m = fmaxf(m, sS[row * LSEQ + col] + mrow[col]);
      }
      for (int off = 1; off < 8; off <<= 1) m = fmaxf(m, __shfl_xor(m, off, 32));
      float s = 0.f;
      for (int i = 0; i < 128; ++i) {
        const int col = seg * 128 + i;
        const float e = __expf(sS[row * LSEQ + col] + mrow[col] - m);
        sS[row * LSEQ + col] = e;
        s += e;
      }
      for (int off = 1; off < 8; off <<= 1) s += __shfl_xor(s, off, 32);
      const float inv = 1.f / s;
      if (hi == 0) {
        for (int i = 0; i < 128; ++i) {
          const int col = seg * 128 + i;
          sD[row * LSEQ + col] = (f16)(sS[row * LSEQ + col] * inv);
        }
      } else {
        for (int i = 0; i < 128; ++i) {
          const int col = seg * 128 + i;
          sD[row * LSEQ + col] =
              (f16)((float)sD[row * LSEQ + col] - lam * sS[row * LSEQ + col] * inv);
        }
      }
    }
    __syncthreads();
  }

  // out(16 x 128) = diff(16 x 1024) @ v(1024 x 128); wave wv -> cols [32wv,32wv+32)
  v8f oacc[2] = {};
  const f16* vh = vt16 + (size_t)(b * HE + he) * 128 * LSEQ;
  for (int ks = 0; ks < 32; ++ks) {
    const int k0 = ks * 32;
    F16Frag fa;
    {
      const int row = lane & 15;
      const int ko = (lane < 16) ? 0 : 8;
      fa.q[0] = *(const uint4*)(sD + row * LSEQ + k0 + ko);
      fa.q[1] = *(const uint4*)(sD + row * LSEQ + k0 + ko + 16);
    }
#pragma unroll
    for (int j = 0; j < 2; ++j) {
      const int c = wv * 32 + j * 16 + (lane & 15);
      const int kbb = (lane < 16) ? 0 : 8;
      F16Frag fb;
      const f16* vb = vh + (size_t)c * LSEQ + k0 + kbb;
      fb.q[0] = *(const uint4*)(vb);
      fb.q[1] = *(const uint4*)(vb + 16);
      oacc[j] = __builtin_amdgcn_wmma_f32_16x16x32_f16(false, fa.v, false, fb.v, (short)0, oacc[j], false, false);
    }
  }
#pragma unroll
  for (int j = 0; j < 2; ++j)
#pragma unroll
    for (int r = 0; r < 8; ++r) {
      const int M = r + ((lane >= 16) ? 8 : 0);
      const int c = wv * 32 + j * 16 + (lane & 15);
      sO[M * 128 + c] = oacc[j][r];
    }
  __syncthreads();
  // headwise RMSNorm over 128, * norm_w * (1 - lambda_init)
  if (t < 16) {
    float ss = 0.f;
    for (int i = 0; i < 128; ++i) { const float v = sO[t * 128 + i]; ss += v * v; }
    sR[t] = rsqrtf(ss * (1.0f / 128.0f) + 1e-5f);
  }
  __syncthreads();
  for (int i = 0; i < 16; ++i) {
    const int idx = t * 16 + i;
    const int row = idx >> 7, col = idx & 127;
    const float v = sO[idx] * sR[row] * normw[col] * (1.0f - LAMBDA_INIT);
    attn16[((size_t)(b * LSEQ + q0 + row)) * DM + he * 128 + col] = (f16)v;
  }
}

// ---------------------------------------------------------------------------
extern "C" void kernel_launch(void* const* d_in, const int* in_sizes, int n_in,
                              void* d_out, int out_size, void* d_ws, size_t ws_size,
                              hipStream_t stream) {
  const float* query = (const float*)d_in[0];
  const float* mask  = (const float*)d_in[3];
  const float* Wq = (const float*)d_in[4];  const float* bq = (const float*)d_in[5];
  const float* Wk = (const float*)d_in[6];  const float* bk = (const float*)d_in[7];
  const float* Wv = (const float*)d_in[8];  const float* bv = (const float*)d_in[9];
  const float* Wo = (const float*)d_in[10]; const float* bo = (const float*)d_in[11];
  const float* normw = (const float*)d_in[12];
  const float* lq1 = (const float*)d_in[13]; const float* lk1 = (const float*)d_in[14];
  const float* lq2 = (const float*)d_in[15]; const float* lk2 = (const float*)d_in[16];

  char* ws = (char*)d_ws;
  const size_t MB = 1024u * 1024u;
  f16* x16    = (f16*)(ws + 0);         // 8 MB
  f16* WqT    = (f16*)(ws + 8  * MB);   // 2 MB
  f16* WkT    = (f16*)(ws + 10 * MB);   // 2 MB
  f16* WvT    = (f16*)(ws + 12 * MB);   // 2 MB
  f16* WoT    = (f16*)(ws + 14 * MB);   // 2 MB
  f16* q16    = (f16*)(ws + 16 * MB);   // 8 MB
  f16* k16    = (f16*)(ws + 24 * MB);   // 8 MB
  f16* vt16   = (f16*)(ws + 32 * MB);   // 8 MB
  f16* attn16 = (f16*)(ws + 40 * MB);   // 8 MB
  float* lam  = (float*)(ws + 48 * MB); // 4 B

  cvt_x_kernel<<<16384, 256, 0, stream>>>(query, x16);
  dim3 tb(32, 8), tg(32, 32);
  cvt_wT_kernel<<<tg, tb, 0, stream>>>(Wq, WqT);
  cvt_wT_kernel<<<tg, tb, 0, stream>>>(Wk, WkT);
  cvt_wT_kernel<<<tg, tb, 0, stream>>>(Wv, WvT);
  cvt_wT_kernel<<<tg, tb, 0, stream>>>(Wo, WoT);
  lambda_kernel<<<1, 64, 0, stream>>>(lq1, lk1, lq2, lk2, lam);

  dim3 gg(16, 64);   // N-tiles x M-tiles
  gemm_kernel<0><<<gg, 128, 0, stream>>>(x16, WqT, bq, (void*)q16);
  gemm_kernel<1><<<gg, 128, 0, stream>>>(x16, WkT, bk, (void*)k16);
  gemm_kernel<2><<<gg, 128, 0, stream>>>(x16, WvT, bv, (void*)vt16);

  const size_t shmem = 16 * 1024 * 4 + 16 * 1024 * 2 + 16 * 128 * 4 + 64;
  attn_kernel<<<2048, 128, shmem, stream>>>(q16, k16, vt16, mask, lam, normw, attn16);

  gemm_kernel<3><<<gg, 128, 0, stream>>>(attn16, WoT, bo, d_out);
}
